// ContextGlobalTransformerDecoder_69346541961852
// MI455X (gfx1250) — matrix operand order repaired and checked by reading, not compiled, not run on hardware
//
#include <hip/hip_runtime.h>
#include <math.h>

// ---------------------------------------------------------------------------
// ContextGlobalTransformerDecoder on gfx1250 (MI455X), wave32 + WMMA bf16.
//
// bf16 operand pipeline: weights converted to bf16 once per call; LN/softmax/
// GEMM emit bf16 operands directly, so the WMMA inner loop is pure b128
// fragment loads + v_wmma_f32_16x16x32_bf16 (no conversion VALU).  B tiles are
// staged to LDS once per 128x64 block (8x reuse across waves) via CDNA5 async
// global->LDS loads (ASYNCcnt).  Residual streams stay fp32.  ~210 MB ws.
// ---------------------------------------------------------------------------

typedef __attribute__((ext_vector_type(16))) __bf16 v16bf;
typedef __attribute__((ext_vector_type(8)))  __bf16 v8bf;
typedef __attribute__((ext_vector_type(8)))  float  v8f;

#if defined(__has_builtin)
# if __has_builtin(__builtin_amdgcn_global_load_async_to_lds_b128)
#  define HAVE_ASYNC_LDS 1
# endif
#endif
#ifndef HAVE_ASYNC_LDS
# define HAVE_ASYNC_LDS 0
#endif

#if HAVE_ASYNC_LDS
// builtin expects (global int4*, shared int4*, imm offset, imm cpol)
typedef int v4i_t __attribute__((vector_size(16)));
typedef __attribute__((address_space(1))) v4i_t v4i_as1;
typedef __attribute__((address_space(3))) v4i_t v4i_as3;
#endif

#define T_TOK  4608          // 8 * 576 tokens
#define DMODEL 512
#define HDIM   64
#define DMLP   2048
#define DIN    1024
#define DOUT   128
#define SMAX   2304
#define NHEAD  8
#define BROW   72            // LDS row stride in bf16 (64 + 8 pad, 144B: 16B-mult,
                             // lane stride 36 dwords -> conflict-free b128 reads)

__device__ __forceinline__ __bf16 to_bf16(float f) {
    unsigned u = __builtin_bit_cast(unsigned, f);
    u += 0x7FFFu + ((u >> 16) & 1u);                 // round-to-nearest-even
    unsigned short h = (unsigned short)(u >> 16);
    return __builtin_bit_cast(__bf16, h);
}

__device__ __forceinline__ float gelu_exact(float x) {
    return 0.5f * x * (1.0f + erff(x * 0.70710678118654752f));
}

// ---------------------------------------------------------------------------
// WMMA GEMM:  C[M,N] = epi( alpha * A[M,K](bf16) * B(bf16) + bias )
// B element (k,n) at B + n*bsn + k*bsk  (weights W[N,K]: bsn=K,bsk=1).
// Block: 256 thr = 8 waves stacked in M -> 128x64 tile; B slab (64k x 64n)
// staged in LDS per block.  M%16==0, N%64==0, K%64==0 (holds for this net).
// ---------------------------------------------------------------------------
template <bool BIAS, bool GELU, bool GAMMA, bool RESID, bool OBF>
__global__ __launch_bounds__(256) void wmma_gemm_kernel(
    const __bf16* __restrict__ A, int lda,
    const __bf16* __restrict__ B, long long bsn, long long bsk,
    const float* __restrict__ bias, const float* __restrict__ gamma,
    const float* __restrict__ resid, int ldr,
    void* __restrict__ Cv, int ldc, int M, int N, int K, float alpha)
{
    __shared__ __attribute__((aligned(16))) __bf16 Bs[64 * BROW];

    const int tid  = threadIdx.x;
    const int wave = tid >> 5, lane = tid & 31;
    const int r = lane & 15, hi = lane >> 4;
    const int mBase = blockIdx.y * 128 + wave * 16;
    const int nBase = blockIdx.x * 64;
    const bool active = (mBase + 16 <= M);    // wave-uniform; inactive waves
                                              // still stage B and hit barriers
    v8f acc[4];
    #pragma unroll
    for (int t = 0; t < 4; ++t) acc[t] = (v8f){0.f,0.f,0.f,0.f,0.f,0.f,0.f,0.f};

    const __bf16* Arow = A + (size_t)(mBase + r) * (size_t)lda;

    // staging assignment: thread -> (column sn, 16-elem chunk sq)
    const int sn = tid >> 2;          // 0..63 (column within N tile)
    const int sq = (tid & 3) * 16;    // k offset within 64-slab

    for (int kb = 0; kb < K; kb += 64) {
        __syncthreads();              // previous slab fully consumed
        if (bsk == 1) {               // contiguous-k B: async bulk copy
            const __bf16* gp = B + (size_t)(nBase + sn) * (size_t)bsn + kb + sq;
            __bf16* lp = &Bs[sn * BROW + sq];
#if HAVE_ASYNC_LDS
            __builtin_amdgcn_global_load_async_to_lds_b128(
                (v4i_as1*)(unsigned long long)gp,
                (v4i_as3*)(unsigned)(unsigned long long)lp, 0, 0);
            __builtin_amdgcn_global_load_async_to_lds_b128(
                (v4i_as1*)(unsigned long long)(gp + 8),
                (v4i_as3*)(unsigned)(unsigned long long)(lp + 8), 0, 0);
# if __has_builtin(__builtin_amdgcn_s_wait_asynccnt)
            __builtin_amdgcn_s_wait_asynccnt(0);
# else
            asm volatile("s_wait_asynccnt 0x0" ::: "memory");
# endif
#else
            *(v8bf*)lp       = *(const v8bf*)gp;
            *(v8bf*)(lp + 8) = *(const v8bf*)(gp + 8);
#endif
        } else {                      // strided-k B (A*V): scalar gather
            const __bf16* gp = B + (size_t)(nBase + sn) * (size_t)bsn
                                 + (size_t)(kb + sq) * (size_t)bsk;
            __bf16* lp = &Bs[sn * BROW + sq];
            #pragma unroll
            for (int j = 0; j < 16; ++j) lp[j] = gp[(size_t)j * (size_t)bsk];
        }
        __syncthreads();

        if (active) {
            __builtin_prefetch(Arow + kb + 64, 0, 3);   // near-temporal prefetch
            #pragma unroll
            for (int kk = 0; kk < 2; ++kk) {
                // A frag (ISA 7.12.2): lane r holds row r; elems 0..7 -> k=hi*8+j,
                // elems 8..15 -> k=16+hi*8+j  => two 16B loads
                const __bf16* ap = Arow + kb + kk * 32 + hi * 8;
                v8bf a0 = *(const v8bf*)ap;
                v8bf a1 = *(const v8bf*)(ap + 16);
                v16bf a = __builtin_shufflevector(a0, a1,
                    0,1,2,3,4,5,6,7,8,9,10,11,12,13,14,15);
                // Batch all four B fragments (8 ds_load_b128 in one clause)
                // before issuing WMMAs so LDS latency overlaps WMMA issue.
                v16bf bfrag[4];
                #pragma unroll
                for (int t = 0; t < 4; ++t) {
                    const __bf16* bp = &Bs[(t * 16 + r) * BROW + kk * 32 + hi * 8];
                    v8bf b0 = *(const v8bf*)bp;
                    v8bf b1 = *(const v8bf*)(bp + 16);
                    bfrag[t] = __builtin_shufflevector(b0, b1,
                        0,1,2,3,4,5,6,7,8,9,10,11,12,13,14,15);
                }
                #pragma unroll
                for (int t = 0; t < 4; ++t)
                    acc[t] = __builtin_amdgcn_wmma_f32_16x16x32_bf16(
                                 false, a, false, bfrag[t], (short)0, acc[t],
                                 false, false);
            }
        }
    }
    if (!active) return;

    float*  Cf = (float*)Cv;
    __bf16* Cb = (__bf16*)Cv;
    #pragma unroll
    for (int t = 0; t < 4; ++t) {
        const int col = nBase + t * 16 + r;
        const float bv = BIAS  ? bias[col]  : 0.0f;
        const float gv = GAMMA ? gamma[col] : 1.0f;
        #pragma unroll
        for (int i = 0; i < 8; ++i) {
            const int row = mBase + hi * 8 + i;
            float v = acc[t][i] * alpha;
            if (BIAS)  v += bv;
            if (GELU)  v = gelu_exact(v);
            if (GAMMA) v *= gv;
            if (RESID) v += resid[(size_t)row * (size_t)ldr + col];
            if (OBF) Cb[(size_t)row * (size_t)ldc + col] = to_bf16(v);
            else     Cf[(size_t)row * (size_t)ldc + col] = v;
        }
    }
}

// ---------------------------------------------------------------------------
// fp32 -> bf16 bulk conversion (weights once per call; X/G before out GEMMs)
// ---------------------------------------------------------------------------
__global__ __launch_bounds__(256) void cvt_kernel(
    const float* __restrict__ in, __bf16* __restrict__ out, long long n)
{
    long long i = (long long)blockIdx.x * 256 + threadIdx.x;
    const long long stride = (long long)gridDim.x * 256;
    for (; i < n; i += stride) out[i] = to_bf16(in[i]);
}

// ---------------------------------------------------------------------------
// LayerNorm: fp32 in -> bf16 out (one 256-thread block per row)
// ---------------------------------------------------------------------------
__global__ __launch_bounds__(256) void ln_kernel(
    const float* __restrict__ x, const float* __restrict__ w,
    const float* __restrict__ b, __bf16* __restrict__ o, int D)
{
    __shared__ float r1[256];
    __shared__ float r2[256];
    const size_t base = (size_t)blockIdx.x * (size_t)D;
    float s = 0.f, s2 = 0.f;
    for (int i = threadIdx.x; i < D; i += 256) {
        float v = x[base + i]; s += v; s2 += v * v;
    }
    r1[threadIdx.x] = s; r2[threadIdx.x] = s2; __syncthreads();
    for (int k = 128; k > 0; k >>= 1) {
        if (threadIdx.x < k) {
            r1[threadIdx.x] += r1[threadIdx.x + k];
            r2[threadIdx.x] += r2[threadIdx.x + k];
        }
        __syncthreads();
    }
    const float mean = r1[0] / (float)D;
    const float var  = r2[0] / (float)D - mean * mean;
    const float inv  = rsqrtf(var + 1e-6f);
    for (int i = threadIdx.x; i < D; i += 256)
        o[base + i] = to_bf16((x[base + i] - mean) * inv * w[i] + b[i]);
}

// ---------------------------------------------------------------------------
// qk-norm: in-place LN over 64 on q/k sections of bf16 QKV [tok,1536].
// One wave32 per (token, head, q|k) row; shuffle reductions.
// ---------------------------------------------------------------------------
__global__ __launch_bounds__(256) void qknorm_kernel(
    __bf16* __restrict__ qkv,
    const float* __restrict__ qw, const float* __restrict__ qb,
    const float* __restrict__ kw, const float* __restrict__ kb, int T)
{
    const int wid  = (blockIdx.x * 256 + threadIdx.x) >> 5;
    const int lane = threadIdx.x & 31;
    if (wid >= T * 16) return;
    const int t = wid >> 4, rem = wid & 15, which = rem >> 3, h = rem & 7;
    __bf16* p = qkv + (size_t)t * 1536 + which * 512 + h * 64;
    float v0 = (float)p[lane], v1 = (float)p[lane + 32];
    float s = v0 + v1, s2 = v0 * v0 + v1 * v1;
    for (int i = 16; i >= 1; i >>= 1) {
        s  += __shfl_xor(s,  i, 32);
        s2 += __shfl_xor(s2, i, 32);
    }
    const float mean = s * (1.0f / 64.0f);
    const float var  = s2 * (1.0f / 64.0f) - mean * mean;
    const float inv  = rsqrtf(var + 1e-6f);
    const float* W = which ? kw : qw;
    const float* B = which ? kb : qb;
    p[lane]      = to_bf16((v0 - mean) * inv * W[lane]      + B[lane]);
    p[lane + 32] = to_bf16((v1 - mean) * inv * W[lane + 32] + B[lane + 32]);
}

// ---------------------------------------------------------------------------
// Row softmax on fp32 scores, emitting bf16 probabilities
// ---------------------------------------------------------------------------
__global__ __launch_bounds__(256) void softmax_kernel(
    float* __restrict__ sc, __bf16* __restrict__ scb, int S)
{
    __shared__ float red[256];
    const size_t base = (size_t)blockIdx.x * (size_t)S;
    float m = -3.4e38f;
    for (int i = threadIdx.x; i < S; i += 256) m = fmaxf(m, sc[base + i]);
    red[threadIdx.x] = m; __syncthreads();
    for (int k = 128; k > 0; k >>= 1) {
        if (threadIdx.x < k)
            red[threadIdx.x] = fmaxf(red[threadIdx.x], red[threadIdx.x + k]);
        __syncthreads();
    }
    m = red[0]; __syncthreads();
    float sum = 0.f;
    for (int i = threadIdx.x; i < S; i += 256) {
        float e = __expf(sc[base + i] - m);
        sc[base + i] = e; sum += e;
    }
    red[threadIdx.x] = sum; __syncthreads();
    for (int k = 128; k > 0; k >>= 1) {
        if (threadIdx.x < k) red[threadIdx.x] += red[threadIdx.x + k];
        __syncthreads();
    }
    const float invs = 1.0f / red[0];
    for (int i = threadIdx.x; i < S; i += 256)
        scb[base + i] = to_bf16(sc[base + i] * invs);
}

// ---------------------------------------------------------------------------
// Host-side helpers
// ---------------------------------------------------------------------------
static void gemm(hipStream_t st, const __bf16* A, int lda, const __bf16* B,
                 long long bsn, long long bsk, const float* bias,
                 const float* gamma, const float* resid, int ldr,
                 void* C, int ldc, int M, int N, int K, float alpha,
                 bool gelu, bool obf)
{
    dim3 g(N / 64, (M + 127) / 128, 1);
    const bool bi = (bias != nullptr), ga = (gamma != nullptr),
               re = (resid != nullptr);
#define LK(BI,GE,GA,RE,OB) \
    wmma_gemm_kernel<BI,GE,GA,RE,OB><<<g,256,0,st>>>(A,lda,B,bsn,bsk,bias,gamma,resid,ldr,C,ldc,M,N,K,alpha)
    if      (!bi && !gelu && !ga && !re && !obf) LK(false,false,false,false,false);
    else if (!bi && !gelu && !ga && !re &&  obf) LK(false,false,false,false,true);
    else if ( bi && !gelu && !ga && !re && !obf) LK(true, false,false,false,false);
    else if ( bi && !gelu && !ga && !re &&  obf) LK(true, false,false,false,true);
    else if ( bi &&  gelu && !ga && !re &&  obf) LK(true, true, false,false,true);
    else if ( bi && !gelu && !ga &&  re && !obf) LK(true, false,false,true, false);
    else if ( bi && !gelu &&  ga &&  re && !obf) LK(true, false,true, true, false);
#undef LK
}

static void cvt(hipStream_t st, const float* in, __bf16* out, long long n) {
    long long blocks = (n + 255) / 256;
    if (blocks > 16384) blocks = 16384;
    cvt_kernel<<<(int)blocks, 256, 0, st>>>(in, out, n);
}

// bf16 weight pool: converts each fp32 weight once per call, returns bf16 ptr
struct WPool { __bf16* base; size_t off; hipStream_t st; };
static __bf16* wc(WPool& wp, const float* w, size_t n) {
    __bf16* d = wp.base + wp.off;
    wp.off += (n + 7) & ~(size_t)7;     // keep 16B alignment
    cvt(wp.st, w, d, (long long)n);
    return d;
}

// SDPA on interleaved bf16 QKV [tok,1536]; reuses one SMAX^2 score buffer
static void attention(hipStream_t st, const __bf16* QKVb, float* SC,
                      __bf16* SCb, __bf16* Ob, int Bn, int S)
{
    for (int b = 0; b < Bn; ++b) {
        for (int h = 0; h < NHEAD; ++h) {
            const __bf16* Qp = QKVb + (size_t)b * S * 1536 + (size_t)h * HDIM;
            // scores = (Q*HD^-0.5) K^T : B(k,n)=K[n][k] -> bsn=1536, bsk=1 (async path)
            gemm(st, Qp, 1536, Qp + 512, 1536, 1, nullptr, nullptr, nullptr, 0,
                 SC, S, S, S, HDIM, 0.125f, false, false);
            softmax_kernel<<<S, 256, 0, st>>>(SC, SCb, S);
            // O = P V : B(k,n)=V[k][n] -> bsn=1, bsk=1536 (gather staging path)
            gemm(st, SCb, S, Qp + 1024, 1, 1536, nullptr, nullptr, nullptr, 0,
                 Ob + (size_t)b * S * DMODEL + (size_t)h * HDIM, DMODEL,
                 S, HDIM, S, 1.0f, false, true);
        }
    }
}

// ---------------------------------------------------------------------------
// kernel_launch — param leaves in jax pytree order (dict keys sorted):
// blocks[5]x24, global_blocks[6]x18, global_linear_out(w,b), linear_out(w,b),
// projects_x(w,b), projects_y(w,b), then scalars B, N.
// ---------------------------------------------------------------------------
extern "C" void kernel_launch(void* const* d_in, const int* in_sizes, int n_in,
                              void* d_out, int out_size, void* d_ws, size_t ws_size,
                              hipStream_t stream)
{
    (void)in_sizes; (void)out_size; (void)ws_size;
    if (n_in < 240) return;

    const float* hidden  = (const float*)d_in[0];
    const float* context = (const float*)d_in[1];
    const float* const* P = (const float* const*)(d_in + 2);

    //  0 ck.w 1 ck.b 2 cproj.w 3 cproj.b 4 cq.w 5 cq.b 6 cv.w 7 cv.b
    //  8 fc1.w 9 fc1.b 10 fc2.w 11 fc2.b 12 n1.w 13 n1.b 14 n2.w 15 n2.b
    // 16 n3.w 17 n3.b 18 ny.w 19 ny.b 20 proj.w 21 proj.b 22 qkv.w 23 qkv.b
    #define CBP(i, off) P[(i) * 24 + (off)]
    //  0 fc1.w 1 fc1.b 2 fc2.w 3 fc2.b 4 gamma1 5 gamma2 6 kn.w 7 kn.b
    //  8 n1.w 9 n1.b 10 n2.w 11 n2.b 12 proj.w 13 proj.b 14 qn.w 15 qn.b
    // 16 qkv.w 17 qkv.b
    #define GBP(i, off) P[120 + (i) * 18 + (off)]
    const float* glo_w = P[228]; const float* glo_b = P[229];
    const float* lo_w  = P[230]; const float* lo_b  = P[231];
    const float* px_w  = P[232]; const float* px_b  = P[233];
    const float* py_w  = P[234]; const float* py_b  = P[235];

    // ---- workspace layout (fp32 region 16B-aligned, then bf16 region) ----
    float* X  = (float*)d_ws;                      // residual stream  [4608,512]
    float* Y  = X  + (size_t)T_TOK * DMODEL;       // context stream
    float* G  = Y  + (size_t)T_TOK * DMODEL;       // global stream
    float* SC = G  + (size_t)T_TOK * DMODEL;       // scores [2304,2304]
    __bf16* HinB  = (__bf16*)(SC + (size_t)SMAX * SMAX);
    __bf16* CtxB  = HinB  + (size_t)T_TOK * DIN;
    __bf16* TLNb  = CtxB  + (size_t)T_TOK * DIN;   // LN(x) / cvt scratch
    __bf16* TYb   = TLNb  + (size_t)T_TOK * DMODEL;
    __bf16* QKVb  = TYb   + (size_t)T_TOK * DMODEL;   // [4608,1536]
    __bf16* HMLPb = QKVb  + (size_t)T_TOK * 1536;     // [4608,2048]
    __bf16* Ob    = HMLPb + (size_t)T_TOK * DMLP;     // attn out [4608,512]
    __bf16* SCb   = Ob    + (size_t)T_TOK * DMODEL;   // bf16 probs
    WPool wp { SCb + (size_t)SMAX * SMAX, 0, stream };// ~41.1M bf16 of weights
    float* out = (float*)d_out;                       // [4608,128]

    // ---- input projections
    cvt(stream, hidden,  HinB, (long long)T_TOK * DIN);
    cvt(stream, context, CtxB, (long long)T_TOK * DIN);
    gemm(stream, HinB, DIN, wc(wp, px_w, (size_t)DMODEL * DIN), DIN, 1, px_b,
         nullptr, nullptr, 0, X, DMODEL, T_TOK, DMODEL, DIN, 1.0f, false, false);
    gemm(stream, CtxB, DIN, wc(wp, py_w, (size_t)DMODEL * DIN), DIN, 1, py_b,
         nullptr, nullptr, 0, Y, DMODEL, T_TOK, DMODEL, DIN, 1.0f, false, false);

    // ---- 5 cross blocks (batch 8, seq 576)
    for (int cb = 0; cb < 5; ++cb) {
        // self-attention
        ln_kernel<<<T_TOK, 256, 0, stream>>>(X, CBP(cb,12), CBP(cb,13), TLNb, DMODEL);
        gemm(stream, TLNb, DMODEL, wc(wp, CBP(cb,22), (size_t)1536 * DMODEL),
             DMODEL, 1, CBP(cb,23), nullptr, nullptr, 0,
             QKVb, 1536, T_TOK, 1536, DMODEL, 1.0f, false, true);
        attention(stream, QKVb, SC, SCb, Ob, 8, 576);
        gemm(stream, Ob, DMODEL, wc(wp, CBP(cb,20), (size_t)DMODEL * DMODEL),
             DMODEL, 1, CBP(cb,21), nullptr, X, DMODEL,
             X, DMODEL, T_TOK, DMODEL, DMODEL, 1.0f, false, false);
        // cross-attention: q from LN(x,norm2); k,v from LN(y,norm_y)
        ln_kernel<<<T_TOK, 256, 0, stream>>>(Y, CBP(cb,18), CBP(cb,19), TYb, DMODEL);
        ln_kernel<<<T_TOK, 256, 0, stream>>>(X, CBP(cb,14), CBP(cb,15), TLNb, DMODEL);
        gemm(stream, TLNb, DMODEL, wc(wp, CBP(cb,4), (size_t)DMODEL * DMODEL),
             DMODEL, 1, CBP(cb,5), nullptr, nullptr, 0,
             QKVb + 0, 1536, T_TOK, DMODEL, DMODEL, 1.0f, false, true);   // q
        gemm(stream, TYb, DMODEL, wc(wp, CBP(cb,0), (size_t)DMODEL * DMODEL),
             DMODEL, 1, CBP(cb,1), nullptr, nullptr, 0,
             QKVb + 512, 1536, T_TOK, DMODEL, DMODEL, 1.0f, false, true); // k
        gemm(stream, TYb, DMODEL, wc(wp, CBP(cb,6), (size_t)DMODEL * DMODEL),
             DMODEL, 1, CBP(cb,7), nullptr, nullptr, 0,
             QKVb + 1024, 1536, T_TOK, DMODEL, DMODEL, 1.0f, false, true);// v
        attention(stream, QKVb, SC, SCb, Ob, 8, 576);
        gemm(stream, Ob, DMODEL, wc(wp, CBP(cb,2), (size_t)DMODEL * DMODEL),
             DMODEL, 1, CBP(cb,3), nullptr, X, DMODEL,
             X, DMODEL, T_TOK, DMODEL, DMODEL, 1.0f, false, false);
        // MLP
        ln_kernel<<<T_TOK, 256, 0, stream>>>(X, CBP(cb,16), CBP(cb,17), TLNb, DMODEL);
        gemm(stream, TLNb, DMODEL, wc(wp, CBP(cb,8), (size_t)DMLP * DMODEL),
             DMODEL, 1, CBP(cb,9), nullptr, nullptr, 0,
             HMLPb, DMLP, T_TOK, DMLP, DMODEL, 1.0f, true, true);         // GELU
        gemm(stream, HMLPb, DMLP, wc(wp, CBP(cb,10), (size_t)DMODEL * DMLP),
             DMLP, 1, CBP(cb,11), nullptr, X, DMODEL,
             X, DMODEL, T_TOK, DMODEL, DMLP, 1.0f, false, false);
    }

    // ---- out = linear_out(X)
    cvt(stream, X, TLNb, (long long)T_TOK * DMODEL);
    gemm(stream, TLNb, DMODEL, wc(wp, lo_w, (size_t)DOUT * DMODEL), DMODEL, 1,
         lo_b, nullptr, nullptr, 0, out, DOUT, T_TOK, DOUT, DMODEL, 1.0f,
         false, false);

    // ---- global stream: g = X; 6 qk-normed layer-scaled blocks alternating
    //      reshape (2,2304)/(8,576) — contiguous, only attention batching changes
    (void)hipMemcpyAsync(G, X, (size_t)T_TOK * DMODEL * sizeof(float),
                         hipMemcpyDeviceToDevice, stream);
    for (int gb = 0; gb < 6; ++gb) {
        const int Bg = (gb % 2 == 0) ? 2 : 8;
        const int Sg = (gb % 2 == 0) ? SMAX : 576;
        ln_kernel<<<T_TOK, 256, 0, stream>>>(G, GBP(gb,8), GBP(gb,9), TLNb, DMODEL);
        gemm(stream, TLNb, DMODEL, wc(wp, GBP(gb,16), (size_t)1536 * DMODEL),
             DMODEL, 1, GBP(gb,17), nullptr, nullptr, 0,
             QKVb, 1536, T_TOK, 1536, DMODEL, 1.0f, false, true);
        qknorm_kernel<<<(T_TOK * 16 + 7) / 8, 256, 0, stream>>>(
            QKVb, GBP(gb,14), GBP(gb,15), GBP(gb,6), GBP(gb,7), T_TOK);
        attention(stream, QKVb, SC, SCb, Ob, Bg, Sg);
        gemm(stream, Ob, DMODEL, wc(wp, GBP(gb,12), (size_t)DMODEL * DMODEL),
             DMODEL, 1, GBP(gb,13), GBP(gb,4), G, DMODEL,
             G, DMODEL, T_TOK, DMODEL, DMODEL, 1.0f, false, false); // +g1*attn
        ln_kernel<<<T_TOK, 256, 0, stream>>>(G, GBP(gb,10), GBP(gb,11), TLNb, DMODEL);
        gemm(stream, TLNb, DMODEL, wc(wp, GBP(gb,0), (size_t)DMLP * DMODEL),
             DMODEL, 1, GBP(gb,1), nullptr, nullptr, 0,
             HMLPb, DMLP, T_TOK, DMLP, DMODEL, 1.0f, true, true);
        gemm(stream, HMLPb, DMLP, wc(wp, GBP(gb,2), (size_t)DMODEL * DMLP),
             DMLP, 1, GBP(gb,3), GBP(gb,5), G, DMODEL,
             G, DMODEL, T_TOK, DMODEL, DMLP, 1.0f, false, false);   // +g2*mlp
    }

    // ---- out += global_linear_out(G)
    cvt(stream, G, TLNb, (long long)T_TOK * DMODEL);
    gemm(stream, TLNb, DMODEL, wc(wp, glo_w, (size_t)DOUT * DMODEL), DMODEL, 1,
         glo_b, nullptr, out, DOUT, out, DOUT, T_TOK, DOUT, DMODEL, 1.0f,
         false, false);
}
